// TH_55233279426966
// MI455X (gfx1250) — matrix operand order repaired and checked by reading, not compiled, not run on hardware
//
#include <hip/hip_runtime.h>
#include <hip/hip_bf16.h>

// ---------------------------------------------------------------------------
// Problem constants: B=4, C=64, W=H=256, P=8 -> N=4096 patches, S=64, D=4096
// ---------------------------------------------------------------------------
#define NPATCH 4096
#define DDIM   4096
#define NELEM  16777216   // B*C*W*H == NPATCH*DDIM

typedef __attribute__((ext_vector_type(16))) _Float16 v16h;
typedef __attribute__((ext_vector_type(8)))  _Float16 v8h;
typedef __attribute__((ext_vector_type(8)))  float    v8f;
typedef __attribute__((ext_vector_type(4)))  float    v4f;
typedef __attribute__((ext_vector_type(4)))  int      v4i;

// Workspace layout (bytes)
#define WS_XF   ((size_t)0)                       // xf f16   [N][D]   32 MB
#define WS_WVB  ((size_t)33554432)                // wv_ f16  [D][D]   32 MB
#define WS_WQ   ((size_t)67108864)                // wq f16   [64][64]  8 KB
#define WS_WK   ((size_t)67108864 + 8192)
#define WS_WV   ((size_t)67108864 + 16384)
#define WS_V    ((size_t)67108864 + 24576)        // v_ f16   [N][D]   32 MB
#define WS_OUT1 (WS_V + (size_t)33554432)         // out1 f16 [N][D]   32 MB

// ---------------------------------------------------------------------------
// CDNA5 async global->LDS copy (ASYNCcnt-tracked). The gfx1250 builtin takes
// (v4i addrspace(1)*, v4i addrspace(3)*, imm offset, imm cpol) — one int4 per
// lane = b128. Guarded so the file still compiles (register-staged fallback)
// on toolchains without the builtin.
// ---------------------------------------------------------------------------
#if __has_builtin(__builtin_amdgcn_global_load_async_to_lds_b128)
#define TH_ASYNC_LDS 1
#else
#define TH_ASYNC_LDS 0
#endif

__device__ __forceinline__ void cp16_g2l(const _Float16* g, _Float16* l) {
#if TH_ASYNC_LDS
  __builtin_amdgcn_global_load_async_to_lds_b128(
      (__attribute__((address_space(1))) v4i*)g,
      (__attribute__((address_space(3))) v4i*)l, 0, 0);
#else
  *(v4f*)l = *(const v4f*)g;
#endif
}

__device__ __forceinline__ void wait_async_done() {
#if TH_ASYNC_LDS
#if __has_builtin(__builtin_amdgcn_s_wait_asynccnt)
  __builtin_amdgcn_s_wait_asynccnt(0);
#else
  asm volatile("s_wait_asynccnt 0" ::: "memory");
#endif
#endif
}

// ---------------------------------------------------------------------------
// WMMA helpers: D = A(16x32 f16) * B(32x16 f16) + C(f32), wave32.
// A frag (ISA 16-bit A 16x32): lane<16 -> M=lane, K {0..7,16..23};
//                              lane>=16 -> M=lane-16, K {8..15,24..31}.
// B frag: lane holds column N=lane%16, same K order as A frag.
// Sources are "K-contiguous row major" arrays -> per-lane 16B/32B vector loads.
// ---------------------------------------------------------------------------
__device__ __forceinline__ v8f wmma_f16f32(v16h a, v16h b, v8f c) {
  return __builtin_amdgcn_wmma_f32_16x16x32_f16(false, a, false, b,
                                                (short)0, c, false, false);
}

__device__ __forceinline__ v16h load_a_frag(const _Float16* base, int ld) {
  const int lane = threadIdx.x & 31;
  const int row  = lane & 15;
  const int koff = (lane >> 4) << 3;   // 0 or 8
  union { v16h v; v4f f[2]; } u;
  const _Float16* p = base + row * ld + koff;
  u.f[0] = *(const v4f*)(p);
  u.f[1] = *(const v4f*)(p + 16);
  return u.v;
}

__device__ __forceinline__ v16h load_b_frag(const _Float16* base, int ld) {
  const int lane = threadIdx.x & 31;
  const int col  = lane & 15;
  const int kb   = (lane >> 4) << 3;   // 0 or 8
  union { v16h v; v4f f[2]; } u;
  const _Float16* p = base + col * ld + kb;
  u.f[0] = *(const v4f*)(p);
  u.f[1] = *(const v4f*)(p + 16);
  return u.v;
}

// ---------------------------------------------------------------------------
// Vectorized fp32 -> fp16 convert (n must be a multiple of 2048; it is)
// ---------------------------------------------------------------------------
__global__ __launch_bounds__(256) void TH_cvt8_f16(const float* __restrict__ src,
                                                   _Float16* __restrict__ dst) {
  int i8 = (blockIdx.x * 256 + threadIdx.x) * 8;
  v4f f0 = *(const v4f*)(src + i8);
  v4f f1 = *(const v4f*)(src + i8 + 4);
  v8h h;
#pragma unroll
  for (int j = 0; j < 4; ++j) {
    h[j]     = (_Float16)f0[j];
    h[j + 4] = (_Float16)f1[j];
  }
  *(v8h*)(dst + i8) = h;
}

// ---------------------------------------------------------------------------
// Patchify: xf[n][c*64 + px*8 + py] = x[b][c][ix*8+px][iy*8+py], f32->f16
//           n = (b*32 + ix)*32 + iy.  8 consecutive py are contiguous in BOTH
//           layouts -> full b128 traffic on read and write.
// ---------------------------------------------------------------------------
__global__ __launch_bounds__(256) void TH_patchify(const float* __restrict__ x,
                                                   _Float16* __restrict__ xf) {
  int i8 = (blockIdx.x * 256 + threadIdx.x) * 8;  // i8 = n*4096 + d, py = 0..7
  int d = i8 & 4095;
  int n = i8 >> 12;
  int px = (d >> 3) & 7, c = d >> 6;
  int iy = n & 31, ix = (n >> 5) & 31, b = n >> 10;
  const float* src = x + (((b * 64 + c) * 256 + ix * 8 + px) * 256 + iy * 8);
  v4f f0 = *(const v4f*)(src);
  v4f f1 = *(const v4f*)(src + 4);
  v8h h;
#pragma unroll
  for (int j = 0; j < 4; ++j) {
    h[j]     = (_Float16)f0[j];
    h[j + 4] = (_Float16)f1[j];
  }
  *(v8h*)(xf + i8) = h;
}

// ---------------------------------------------------------------------------
// Intra-patch attention: one workgroup (8 wave32) per patch.
//   q,k,v = W * xp  (+bias);  E = softmax_rows(q^T k);  out1 = v * E^T
// All tiles 64x64, fully LDS resident; 5 WMMA GEMM stages.
// ---------------------------------------------------------------------------
__global__ __launch_bounds__(256) void TH_intra_attn(
    const _Float16* __restrict__ xf,
    const _Float16* __restrict__ wq, const float* __restrict__ bq,
    const _Float16* __restrict__ wk, const float* __restrict__ bk,
    const _Float16* __restrict__ wv, const float* __restrict__ bv,
    _Float16* __restrict__ out1) {
  __shared__ __align__(16) _Float16 xpT[64][64];   // [s][c]
  __shared__ __align__(16) _Float16 qT [64][64];   // [s][o]
  __shared__ __align__(16) _Float16 kT [64][64];   // [t][o]
  __shared__ __align__(16) _Float16 vS [64][64];   // [o][s]
  __shared__ __align__(16) float    eF [64][64];   // [s][t] f32 pre-softmax
  __shared__ __align__(16) _Float16 eH [64][64];   // [i][j] softmaxed f16

  const int n    = blockIdx.x;
  const int tid  = threadIdx.x;
  const int lane = tid & 31;
  const int wid  = tid >> 5;          // 0..7
  const int wm   = wid >> 2;          // 0..1 (row half)
  const int wn   = wid & 3;           // 0..3 (col tile)
  const int col  = lane & 15;
  const int hi   = lane >> 4;

  // Stage xp transposed: xpT[s][c] = xf[n][c*64+s]
  const _Float16* xsrc = xf + (size_t)n * DDIM;
  for (int i8 = tid * 8; i8 < DDIM; i8 += 256 * 8) {
    int c = i8 >> 6;
    int s = i8 & 63;
    v8h t = *(const v8h*)(xsrc + i8);
#pragma unroll
    for (int j = 0; j < 8; ++j) xpT[s + j][c] = t[j];
  }
  __syncthreads();

  const v8f vzero = {0.f, 0.f, 0.f, 0.f, 0.f, 0.f, 0.f, 0.f};

  // q / k / v GEMMs (A = weights from global/L2, B = xpT in LDS)
#pragma unroll
  for (int tm = 0; tm < 2; ++tm) {
    const int m0 = (wm * 2 + tm) * 16;   // output channel tile
    const int n0 = wn * 16;              // spatial tile
    v8f aq = vzero, ak = vzero, av = vzero;
#pragma unroll
    for (int k0 = 0; k0 < 64; k0 += 32) {
      v16h bfrag = load_b_frag(&xpT[n0][k0], 64);
      v16h aw;
      aw = load_a_frag(wq + m0 * 64 + k0, 64); aq = wmma_f16f32(aw, bfrag, aq);
      aw = load_a_frag(wk + m0 * 64 + k0, 64); ak = wmma_f16f32(aw, bfrag, ak);
      aw = load_a_frag(wv + m0 * 64 + k0, 64); av = wmma_f16f32(aw, bfrag, av);
    }
    const int s  = n0 + col;
    const int ob = m0 + hi * 8;
    v8h hq, hk;
#pragma unroll
    for (int r = 0; r < 8; ++r) {
      hq[r] = (_Float16)(aq[r] + bq[ob + r]);
      hk[r] = (_Float16)(ak[r] + bk[ob + r]);
      vS[ob + r][s] = (_Float16)(av[r] + bv[ob + r]);
    }
    *(v8h*)(&qT[s][ob]) = hq;   // transposed store: contiguous 16B per lane
    *(v8h*)(&kT[s][ob]) = hk;
  }
  __syncthreads();

  // Energy: E[s][t] = sum_o q[o][s]*k[o][t]  (A = qT, B = kT)
#pragma unroll
  for (int tm = 0; tm < 2; ++tm) {
    const int m0 = (wm * 2 + tm) * 16;
    const int n0 = wn * 16;
    v8f acc = vzero;
#pragma unroll
    for (int k0 = 0; k0 < 64; k0 += 32) {
      v16h a = load_a_frag(&qT[m0][k0], 64);
      v16h b = load_b_frag(&kT[n0][k0], 64);
      acc = wmma_f16f32(a, b, acc);
    }
    const int t = n0 + col;
#pragma unroll
    for (int r = 0; r < 8; ++r) eF[m0 + hi * 8 + r][t] = acc[r];
  }
  __syncthreads();

  // Row softmax over t (64 rows, one thread each)
  if (tid < 64) {
    float m = -3.0e38f;
#pragma unroll 8
    for (int j = 0; j < 64; ++j) m = fmaxf(m, eF[tid][j]);
    float sum = 0.f;
#pragma unroll 8
    for (int j = 0; j < 64; ++j) {
      float e = __expf(eF[tid][j] - m);
      eF[tid][j] = e;
      sum += e;
    }
    float inv = 1.0f / sum;
#pragma unroll 8
    for (int j = 0; j < 64; ++j) eH[tid][j] = (_Float16)(eF[tid][j] * inv);
  }
  __syncthreads();

  // out1[c][i] = sum_j v[c][j] * E[i][j]  (A = vS, B = eH)
  _Float16* dst = out1 + (size_t)n * DDIM;
#pragma unroll
  for (int tm = 0; tm < 2; ++tm) {
    const int m0 = (wm * 2 + tm) * 16;
    const int n0 = wn * 16;
    v8f acc = vzero;
#pragma unroll
    for (int k0 = 0; k0 < 64; k0 += 32) {
      v16h a = load_a_frag(&vS[m0][k0], 64);
      v16h b = load_b_frag(&eH[n0][k0], 64);
      acc = wmma_f16f32(a, b, acc);
    }
    const int i = n0 + col;
#pragma unroll
    for (int r = 0; r < 8; ++r) {
      int c = m0 + hi * 8 + r;
      dst[c * 64 + i] = (_Float16)acc[r];
    }
  }
}

// ---------------------------------------------------------------------------
// Inter-patch branch (live part only): v_[n][o] = xf[n][:] . wv_[o][:] + bv_[o]
// 4096^3 f16 GEMM, f32 accumulate. 128x128 tile per 256-thread WG, 8 waves in
// a 2x4 grid -> 8 WMMA tiles per wave per 32-deep K step. Double-buffered LDS
// (ping-pong) with async global->LDS copies overlapping the WMMA block; one
// barrier per K step. Both operands are 32 MB f16 -> resident in the 192 MB L2.
// ---------------------------------------------------------------------------
__global__ __launch_bounds__(256) void TH_gemm_v(
    const _Float16* __restrict__ A,    // xf  [N][K]
    const _Float16* __restrict__ Bw,   // wv_ [O][K] (f16)
    const float* __restrict__ bias,    // bv_ [O]
    _Float16* __restrict__ C) {        // v_  [N][O]
  __shared__ __align__(16) _Float16 As[2][128][32];
  __shared__ __align__(16) _Float16 Bs[2][128][32];
  const int K  = DDIM;
  const int n0 = blockIdx.y * 128;
  const int o0 = blockIdx.x * 128;
  const int tid  = threadIdx.x;
  const int lane = tid & 31;
  const int wid  = tid >> 5;
  const int wm   = wid >> 2;          // 0..1 -> 64 rows each
  const int wn   = wid & 3;           // 0..3 -> 32 cols each
  const int r    = tid >> 1;          // staging row 0..127
  const int kp   = (tid & 1) * 16;    // staging k-part (halves)

  const _Float16* ag = A  + (size_t)(n0 + r) * K + kp;
  const _Float16* bg = Bw + (size_t)(o0 + r) * K + kp;

  const v8f vzero = {0.f, 0.f, 0.f, 0.f, 0.f, 0.f, 0.f, 0.f};
  v8f acc[4][2];
#pragma unroll
  for (int i = 0; i < 4; ++i)
#pragma unroll
    for (int j = 0; j < 2; ++j) acc[i][j] = vzero;

  // Prologue: tile 0 -> buffer 0
  cp16_g2l(ag,     &As[0][r][kp]);
  cp16_g2l(ag + 8, &As[0][r][kp] + 8);
  cp16_g2l(bg,     &Bs[0][r][kp]);
  cp16_g2l(bg + 8, &Bs[0][r][kp] + 8);
  wait_async_done();
  __syncthreads();

  const int NSTEP = K / 32;           // 128
  for (int step = 0; step < NSTEP; ++step) {
    const int cur  = step & 1;
    const int nxt  = cur ^ 1;
    const bool more = (step + 1) < NSTEP;
    if (more) {
      // Issue next tile's copies BEFORE the WMMA block; the buffer being
      // overwritten was last read at step-1 and fenced by that step's barrier.
      const int ko = (step + 1) * 32;
      cp16_g2l(ag + ko,     &As[nxt][r][kp]);
      cp16_g2l(ag + ko + 8, &As[nxt][r][kp] + 8);
      cp16_g2l(bg + ko,     &Bs[nxt][r][kp]);
      cp16_g2l(bg + ko + 8, &Bs[nxt][r][kp] + 8);
      if (step + 2 < NSTEP) {         // hint step+2 toward L2/WGP$
        __builtin_prefetch(ag + ko + 32, 0, 1);
        __builtin_prefetch(bg + ko + 32, 0, 1);
      }
    }

    v16h af[4], bf[2];
#pragma unroll
    for (int mt = 0; mt < 4; ++mt)
      af[mt] = load_a_frag(&As[cur][wm * 64 + mt * 16][0], 32);
#pragma unroll
    for (int nt = 0; nt < 2; ++nt)
      bf[nt] = load_b_frag(&Bs[cur][wn * 32 + nt * 16][0], 32);
#pragma unroll
    for (int mt = 0; mt < 4; ++mt)
#pragma unroll
      for (int nt = 0; nt < 2; ++nt)
        acc[mt][nt] = wmma_f16f32(af[mt], bf[nt], acc[mt][nt]);

    if (more) wait_async_done();      // async writes landed before the fence
    __syncthreads();
  }

  const int col = lane & 15;
  const int hi  = lane >> 4;
#pragma unroll
  for (int nt = 0; nt < 2; ++nt) {
    const int o  = o0 + wn * 32 + nt * 16 + col;
    const float bb = bias[o];
#pragma unroll
    for (int mt = 0; mt < 4; ++mt) {
      const int nrow = n0 + wm * 64 + mt * 16 + hi * 8;
#pragma unroll
      for (int rr = 0; rr < 8; ++rr)
        C[(size_t)(nrow + rr) * DDIM + o] = (_Float16)(acc[mt][nt][rr] + bb);
    }
  }
}

// ---------------------------------------------------------------------------
// Final gate: out = x + x * sigmoid(res1 + res2), un-patching on the fly.
// softmax over the length-1 axis == 1.0 exactly, so res2 = v_ (q_/k_ dead).
// 8 consecutive hy share one (n,d)-run -> full vector traffic everywhere.
// ---------------------------------------------------------------------------
__global__ __launch_bounds__(256) void TH_finalize(
    const float* __restrict__ x,
    const _Float16* __restrict__ out1,   // [N][D]
    const _Float16* __restrict__ v2,     // [N][D]
    float* __restrict__ out) {
  int i8 = (blockIdx.x * 256 + threadIdx.x) * 8;  // x layout [b][c][wx][hy]
  int hy = i8 & 255;                              // multiple of 8 -> py = 0..7
  int wx = (i8 >> 8) & 255;
  int c  = (i8 >> 16) & 63;
  int b  = i8 >> 22;
  int ix = wx >> 3, px = wx & 7;
  int iy = hy >> 3;
  int n = (b * 32 + ix) * 32 + iy;
  int d = (c << 6) | (px << 3);
  size_t pd = (size_t)n * DDIM + d;
  v8h r1 = *(const v8h*)(out1 + pd);
  v8h r2 = *(const v8h*)(v2 + pd);
  v4f x0 = *(const v4f*)(x + i8);
  v4f x1 = *(const v4f*)(x + i8 + 4);
  v4f o0, o1;
#pragma unroll
  for (int j = 0; j < 4; ++j) {
    float s0 = (float)r1[j] + (float)r2[j];
    float s1 = (float)r1[j + 4] + (float)r2[j + 4];
    float g0 = 1.0f / (1.0f + __expf(-s0));
    float g1 = 1.0f / (1.0f + __expf(-s1));
    o0[j] = x0[j] + x0[j] * g0;
    o1[j] = x1[j] + x1[j] * g1;
  }
  *(v4f*)(out + i8)     = o0;
  *(v4f*)(out + i8 + 4) = o1;
}

// ---------------------------------------------------------------------------
// Launch
// ---------------------------------------------------------------------------
extern "C" void kernel_launch(void* const* d_in, const int* in_sizes, int n_in,
                              void* d_out, int out_size, void* d_ws, size_t ws_size,
                              hipStream_t stream) {
  (void)in_sizes; (void)n_in; (void)out_size; (void)ws_size;
  const float* x   = (const float*)d_in[0];
  const float* wq  = (const float*)d_in[1];
  const float* bq  = (const float*)d_in[2];
  const float* wk  = (const float*)d_in[3];
  const float* bk  = (const float*)d_in[4];
  const float* wvv = (const float*)d_in[5];
  const float* bv  = (const float*)d_in[6];
  // d_in[7..10] = wq_, bq_, wk_, bk_ : mathematically dead
  // (softmax over an axis of length 1 is identically 1.0)
  const float* wvb = (const float*)d_in[11];
  const float* bvb = (const float*)d_in[12];
  float* out = (float*)d_out;

  char* ws = (char*)d_ws;
  _Float16* xf_h   = (_Float16*)(ws + WS_XF);
  _Float16* wvb_h  = (_Float16*)(ws + WS_WVB);
  _Float16* wq_h   = (_Float16*)(ws + WS_WQ);
  _Float16* wk_h   = (_Float16*)(ws + WS_WK);
  _Float16* wv_h   = (_Float16*)(ws + WS_WV);
  _Float16* v_h    = (_Float16*)(ws + WS_V);
  _Float16* out1_h = (_Float16*)(ws + WS_OUT1);

  // Precision staging (wv_ as f16 is 32 MB -> L2-resident for the big GEMM)
  TH_cvt8_f16<<<NELEM / 2048, 256, 0, stream>>>(wvb, wvb_h);
  TH_cvt8_f16<<<2, 256, 0, stream>>>(wq,  wq_h);
  TH_cvt8_f16<<<2, 256, 0, stream>>>(wk,  wk_h);
  TH_cvt8_f16<<<2, 256, 0, stream>>>(wvv, wv_h);

  TH_patchify<<<NELEM / 2048, 256, 0, stream>>>(x, xf_h);

  TH_intra_attn<<<NPATCH, 256, 0, stream>>>(xf_h, wq_h, bq, wk_h, bk, wv_h, bv,
                                            out1_h);

  TH_gemm_v<<<dim3(32, 32), 256, 0, stream>>>(xf_h, wvb_h, bvb, v_h);

  TH_finalize<<<NELEM / 2048, 256, 0, stream>>>(x, out1_h, v_h, out);
}
